// GridLSTM_88751204205227
// MI455X (gfx1250) — matrix operand order repaired and checked by reading, not compiled, not run on hardware
//
#include <hip/hip_runtime.h>
#include <hip/hip_bf16.h>
#include <math.h>

typedef __attribute__((ext_vector_type(16))) __bf16 v16bf;
typedef __attribute__((ext_vector_type(8)))  float  v8f;

#define B_    128
#define T_    256
#define L_    2
#define H_    512
#define G_    2048   /* 4*H */
#define K_    1024   /* 2*H */
#define KC    32
#define NK    (K_ / KC)
#define NBLK  32
#define NTHR  256

/* packed frag-major weight layout strides (bf16 elements) */
#define WF_PER_KT   512                       /* 32 lanes * 16 elems */
#define WF_PER_NT   ((K_ / KC) * WF_PER_KT)   /* 16384 */
#define WF_PER_CELL ((G_ / 16) * WF_PER_NT)   /* 2097152 */
#define WF_TOTAL    ((size_t)4 * WF_PER_CELL) /* 8388608 elems = 16 MB */

/* flat float offsets into d_out: (H0_out, H1f, M0_out, M1f) */
#define H0OUT ((size_t)0)
#define H1F   ((size_t)16777216)            /* B*T*H */
#define M0OUT ((size_t)16908288)            /* + B*L*H */
#define M1F   ((size_t)33685504)            /* + B*T*H */

__device__ __forceinline__ unsigned short f2bf(float f) {
  unsigned u = __float_as_uint(f);
  unsigned r = 0x7FFFu + ((u >> 16) & 1u);   /* round-to-nearest-even */
  return (unsigned short)((u + r) >> 16);
}
__device__ __forceinline__ float sigf(float x) { return 1.0f / (1.0f + __expf(-x)); }
__device__ __forceinline__ float tanh_fast(float x) {
  float t = __expf(-2.0f * fabsf(x));
  float r = (1.0f - t) / (1.0f + t);
  return copysignf(r, x);
}

/* CDNA5 async global->LDS copy, 16B per lane (tracked by ASYNCcnt) */
__device__ __forceinline__ void async_lds_b128(unsigned lds_off, const void* gaddr) {
  asm volatile("global_load_async_to_lds_b128 %0, %1, off"
               :: "v"(lds_off), "v"((unsigned long long)gaddr)
               : "memory");
}
__device__ __forceinline__ void wait_async0() {
  asm volatile("s_wait_asynccnt 0x0" ::: "memory");
}

/* generation-based grid barrier: cnt reset BEFORE gen release -> race-free */
__device__ __forceinline__ void gbar(unsigned* cnt, unsigned* gen) {
  __syncthreads();
  if (threadIdx.x == 0) {
    unsigned g = __hip_atomic_load(gen, __ATOMIC_RELAXED, __HIP_MEMORY_SCOPE_AGENT);
    unsigned prev = __hip_atomic_fetch_add(cnt, 1u, __ATOMIC_ACQ_REL, __HIP_MEMORY_SCOPE_AGENT);
    if (prev == (unsigned)(NBLK - 1)) {
      __hip_atomic_store(cnt, 0u, __ATOMIC_RELAXED, __HIP_MEMORY_SCOPE_AGENT);
      __hip_atomic_fetch_add(gen, 1u, __ATOMIC_RELEASE, __HIP_MEMORY_SCOPE_AGENT);
    } else {
      while (__hip_atomic_load(gen, __ATOMIC_ACQUIRE, __HIP_MEMORY_SCOPE_AGENT) == g)
        __builtin_amdgcn_s_sleep(1);
    }
  }
  __syncthreads();
}

__global__ void grid_lstm_prep(const float* __restrict__ H0in,
                               const float* __restrict__ Wih,
                               const float* __restrict__ Whh,
                               const float* __restrict__ bih,
                               const float* __restrict__ bhh,
                               float* __restrict__ bias,
                               unsigned short* __restrict__ Wf,
                               float* __restrict__ m1s,
                               unsigned short* __restrict__ h1b,
                               unsigned short* __restrict__ h0I,
                               unsigned* __restrict__ barcnt,
                               int do_pack)
{
  const size_t gid = (size_t)blockIdx.x * blockDim.x + threadIdx.x;
  const size_t stride = (size_t)gridDim.x * blockDim.x;
  if (gid == 0) *barcnt = 0u;
  /* bias packed with gate-interleave: column n = 4*u + gate <-> row = gate*512+u */
  for (size_t i = gid; i < 4 * G_; i += stride) {
    int n = (int)i & (G_ - 1);
    int cell = (int)(i >> 11);
    int rold = (n & 3) * 512 + (n >> 2);
    bias[i] = bih[cell * G_ + rold] + bhh[cell * G_ + rold];
  }
  for (size_t i = gid; i < (size_t)L_ * B_ * H_; i += stride) m1s[i] = 0.0f;
  for (size_t i = gid; i < (size_t)4 * B_ * H_; i += stride) h1b[i] = 0;
  /* t=0 step input -> bf16, parity-0 buffer */
  for (size_t i = gid; i < (size_t)B_ * H_; i += stride) {
    int m = (int)(i >> 9), k = (int)i & (H_ - 1);
    h0I[i] = f2bf(H0in[(size_t)m * (T_ * H_) + k]);
  }
  /* pack weights into WMMA-B-fragment-major bf16 layout */
  if (do_pack) {
    for (size_t i = gid; i < WF_TOTAL; i += stride) {
      int e    = (int)i & 15;
      int lane = (int)(i >> 4) & 31;
      int kt   = (int)(i >> 9) & 31;
      int nt   = (int)(i >> 14) & 127;
      int cell = (int)(i >> 21);
      int n  = nt * 16 + (lane & 15);
      int kk = kt * KC + (lane >> 4) * 16 + e;
      size_t rbase = ((size_t)cell * G_ + (size_t)(n & 3) * 512 + (n >> 2)) * H_;
      float w = (kk < H_) ? Wih[rbase + kk] : Whh[rbase + (kk - H_)];
      Wf[i] = f2bf(w);
    }
  }
}

template <bool PRE>
__global__ __launch_bounds__(NTHR, 1)
void grid_lstm_main(const float* __restrict__ H0in,
                    const float* __restrict__ Wih,
                    const float* __restrict__ Whh,
                    float* __restrict__ out,
                    const float* __restrict__ bias,
                    const unsigned short* __restrict__ Wf,
                    float* m0s,
                    float* m1s,                 /* [2][B*H] */
                    unsigned short* h0I,        /* [2][B*H] step-input double buffer */
                    unsigned short* h0_mid,     /* cell0 output */
                    unsigned short* h0_fin,     /* cell2 output */
                    unsigned short* h1b,        /* [layer][parity][B*H] */
                    unsigned* barcnt, unsigned* bargen)
{
  __shared__ __align__(16) unsigned short As[2][64 * KC];  /* 8 KB, double buffered */
  __shared__ __align__(16) unsigned short Ws[128 * KC];    /* 8 KB (fallback path) */
  __shared__ float gst[8][32][33];                         /* 33 KB gate staging */

  const int tid  = threadIdx.x;
  const int blk  = blockIdx.x;
  const int wave = tid >> 5;
  const int lane = tid & 31;
  const int mb = blk & 1, nb = blk >> 1;
  const int m0 = mb * 64, n0 = nb * 128;
  const int wm = wave & 1, wn = wave >> 1;        /* 2x4 waves -> 64x128 block tile */

  /* invariant per-thread staging coords */
  const int arow = tid >> 2, aseg = tid & 3;      /* A: 64 rows x 4 chunks of 8 bf16 */
  const int wrow = tid >> 1, whalf = tid & 1;     /* fallback W: 128 cols x 2 chunks */
  const int ntile0 = (n0 + wn * 32) >> 4;         /* this wave's first B 16-col tile */

  /* LDS byte offsets of this thread's A staging slot, both buffers */
  const unsigned ldsA0 = (unsigned)(size_t)&As[0][arow * KC + aseg * 8];
  const unsigned ldsA1 = (unsigned)(size_t)&As[1][arow * KC + aseg * 8];

  for (int t = 0; t < T_; ++t) {
    const int p = t & 1;
    for (int cell = 0; cell < 4; ++cell) {
      const int l = cell >> 1, d = cell & 1;
      const unsigned short* Ax;  const unsigned short* Ah;
      float* cst;  unsigned short* hb;
      if (cell == 0)      { Ax = h1b + (size_t)(0 + p) * (B_ * H_); Ah = h0I + (size_t)p * (B_ * H_);
                            cst = m0s;            hb = h0_mid; }
      else if (cell == 1) { Ax = h0_mid; Ah = h1b + (size_t)(0 + p) * (B_ * H_);
                            cst = m1s;            hb = h1b + (size_t)(0 + (1 - p)) * (B_ * H_); }
      else if (cell == 2) { Ax = h1b + (size_t)(2 + p) * (B_ * H_); Ah = h0_mid;
                            cst = m0s;            hb = h0_fin; }
      else                { Ax = h0_fin;  Ah = h1b + (size_t)(2 + p) * (B_ * H_);
                            cst = m1s + B_ * H_;  hb = h1b + (size_t)(2 + (1 - p)) * (B_ * H_); }

      /* fragment-major packed weight pointers (PRE path) */
      const unsigned short* wf0 =
          Wf + (size_t)cell * WF_PER_CELL + (size_t)ntile0 * WF_PER_NT + lane * 16;
      const unsigned short* wf1 = wf0 + WF_PER_NT;

      /* fallback path: gate-interleaved fp32 W row for this thread's column */
      const float* wih = Wih + (size_t)cell * G_ * H_;
      const float* whh = Whh + (size_t)cell * G_ * H_;
      const int ng = n0 + wrow;
      const size_t wbase = (size_t)((ng & 3) * 512 + (ng >> 2)) * H_;

      v8f acc00 = {}, acc01 = {}, acc10 = {}, acc11 = {};

      const int akb = (lane >> 4) * 16;       /* A k: {0..15} / {16..31} */
      const int ar0 = wm * 32 + (lane & 15);

      if (PRE) {
        /* software-pipelined: async A tile (global->LDS DMA), one barrier/iter */
        const size_t rowoff = (size_t)(m0 + arow) * H_;
        {
          int gk = aseg * 8;   /* kt = 0 */
          const unsigned short* sp = (gk < H_) ? (Ax + rowoff + gk) : (Ah + rowoff + gk - H_);
          async_lds_b128(ldsA0, sp);
        }
#pragma unroll 1
        for (int kt = 0; kt < NK; ++kt) {
          wait_async0();           /* own async fill of buf[kt&1] complete */
          __syncthreads();         /* publish; prior-iter LDS reads already done */
          if (kt + 1 < NK) {
            int gk = (kt + 1) * KC + aseg * 8;
            const unsigned short* sp = (gk < H_) ? (Ax + rowoff + gk) : (Ah + rowoff + gk - H_);
            async_lds_b128((kt & 1) ? ldsA0 : ldsA1, sp);
          }
          const unsigned short* Ab = As[kt & 1];
          union FA { uint4 q[2]; v16bf v; } a0, a1;
          union FB { uint4 q[2]; v16bf v; } b0, b1;
          a0.q[0] = *(const uint4*)&Ab[(ar0     ) * KC + akb];
          a0.q[1] = *(const uint4*)&Ab[(ar0     ) * KC + akb + 8];
          a1.q[0] = *(const uint4*)&Ab[(ar0 + 16) * KC + akb];
          a1.q[1] = *(const uint4*)&Ab[(ar0 + 16) * KC + akb + 8];
          const unsigned short* q0 = wf0 + (size_t)kt * WF_PER_KT;
          const unsigned short* q1 = wf1 + (size_t)kt * WF_PER_KT;
          b0.q[0] = *(const uint4*)q0;  b0.q[1] = *(const uint4*)(q0 + 8);
          b1.q[0] = *(const uint4*)q1;  b1.q[1] = *(const uint4*)(q1 + 8);
          if (kt + 1 < NK) {
            __builtin_prefetch(q0 + WF_PER_KT, 0, 3);   /* global_prefetch_b8 */
            __builtin_prefetch(q1 + WF_PER_KT, 0, 3);
          }
          acc00 = __builtin_amdgcn_wmma_f32_16x16x32_bf16(false, a0.v, false, b0.v, (short)0, acc00, false, false);
          acc01 = __builtin_amdgcn_wmma_f32_16x16x32_bf16(false, a0.v, false, b1.v, (short)0, acc01, false, false);
          acc10 = __builtin_amdgcn_wmma_f32_16x16x32_bf16(false, a1.v, false, b0.v, (short)0, acc10, false, false);
          acc11 = __builtin_amdgcn_wmma_f32_16x16x32_bf16(false, a1.v, false, b1.v, (short)0, acc11, false, false);
        }
      } else {
        /* fallback: synchronous staging with on-the-fly fp32->bf16 conversion */
#pragma unroll 1
        for (int kt = 0; kt < NK; ++kt) {
          const int k0 = kt * KC;
          __syncthreads();
          {
            int gk = k0 + aseg * 8;
            const unsigned short* sp = (gk < H_)
                ? (Ax + (size_t)(m0 + arow) * H_ + gk)
                : (Ah + (size_t)(m0 + arow) * H_ + (gk - H_));
            *(uint4*)&As[0][arow * KC + aseg * 8] = *(const uint4*)sp;
          }
          {
            int gk = k0 + whalf * 16;
            const float* wp = (gk < H_) ? (wih + wbase + gk) : (whh + wbase + (gk - H_));
            alignas(16) unsigned short tmp[16];
#pragma unroll
            for (int j = 0; j < 16; j += 4) {
              float4 f = *(const float4*)(wp + j);
              tmp[j + 0] = f2bf(f.x); tmp[j + 1] = f2bf(f.y);
              tmp[j + 2] = f2bf(f.z); tmp[j + 3] = f2bf(f.w);
            }
            *(uint4*)&Ws[wrow * KC + whalf * 16 + 0] = *(const uint4*)&tmp[0];
            *(uint4*)&Ws[wrow * KC + whalf * 16 + 8] = *(const uint4*)&tmp[8];
          }
          __syncthreads();
          union FA { uint4 q[2]; v16bf v; } a0, a1;
          union FB { uint4 q[2]; v16bf v; } b0, b1;
          a0.q[0] = *(const uint4*)&As[0][(ar0     ) * KC + akb];
          a0.q[1] = *(const uint4*)&As[0][(ar0     ) * KC + akb + 8];
          a1.q[0] = *(const uint4*)&As[0][(ar0 + 16) * KC + akb];
          a1.q[1] = *(const uint4*)&As[0][(ar0 + 16) * KC + akb + 8];
          const int bkb = (lane >> 4) * 16;
          const int bc0 = wn * 32 + (lane & 15);
          b0.q[0] = *(const uint4*)&Ws[(bc0     ) * KC + bkb];
          b0.q[1] = *(const uint4*)&Ws[(bc0     ) * KC + bkb + 8];
          b1.q[0] = *(const uint4*)&Ws[(bc0 + 16) * KC + bkb];
          b1.q[1] = *(const uint4*)&Ws[(bc0 + 16) * KC + bkb + 8];
          acc00 = __builtin_amdgcn_wmma_f32_16x16x32_bf16(false, a0.v, false, b0.v, (short)0, acc00, false, false);
          acc01 = __builtin_amdgcn_wmma_f32_16x16x32_bf16(false, a0.v, false, b1.v, (short)0, acc01, false, false);
          acc10 = __builtin_amdgcn_wmma_f32_16x16x32_bf16(false, a1.v, false, b0.v, (short)0, acc10, false, false);
          acc11 = __builtin_amdgcn_wmma_f32_16x16x32_bf16(false, a1.v, false, b1.v, (short)0, acc11, false, false);
        }
      }

      /* bias (per output column -> per lane constant) */
      {
        const float* bp = bias + cell * G_;
        float bia0 = bp[n0 + wn * 32 + (lane & 15)];
        float bia1 = bp[n0 + wn * 32 + 16 + (lane & 15)];
        acc00 += bia0; acc01 += bia1; acc10 += bia0; acc11 += bia1;
      }

      /* stage gates through LDS to undo the C-fragment lane layout */
      __syncthreads();
      {
        int r2 = (lane >> 4) * 8;
        int c2 = lane & 15;
#pragma unroll
        for (int v = 0; v < 8; ++v) {
          gst[wave][r2 + v][c2]           = acc00[v];
          gst[wave][r2 + v][16 + c2]      = acc01[v];
          gst[wave][16 + r2 + v][c2]      = acc10[v];
          gst[wave][16 + r2 + v][16 + c2] = acc11[v];
        }
      }
      __syncthreads();

      /* pointwise LSTM update: lane owns one row, 8 hidden units */
      {
        const int mg = m0 + wm * 32 + lane;
        const int ug0 = (n0 + wn * 32) >> 2;
#pragma unroll
        for (int u = 0; u < 8; ++u) {
          float gi = gst[wave][lane][4 * u + 0];
          float gf = gst[wave][lane][4 * u + 1];
          float gg = gst[wave][lane][4 * u + 2];
          float go = gst[wave][lane][4 * u + 3];
          int ug = ug0 + u;
          size_t sidx = (size_t)mg * H_ + ug;
          float cold = (cell == 0) ? 0.0f : cst[sidx];   /* m0 = 0 at step start */
          float cn = sigf(gf) * cold + sigf(gi) * tanh_fast(gg);
          float hn = sigf(go) * tanh_fast(cn);
          cst[sidx] = cn;
          hb[sidx]  = f2bf(hn);
          if (cell == 2) {                       /* h0,m0 after layer loop -> outputs */
            size_t oidx = (size_t)mg * (T_ * H_) + (size_t)t * H_ + ug;
            out[H0OUT + oidx] = hn;
            out[M0OUT + oidx] = cn;
          }
          if (d == 1 && t == T_ - 1) {           /* final H1/M1 states */
            size_t oidx = (size_t)mg * (L_ * H_) + (size_t)l * H_ + ug;
            out[H1F + oidx] = hn;
            out[M1F + oidx] = cn;
          }
        }
      }

      /* fold next step's input conversion into cell0 phase (no extra barrier) */
      if (cell == 0 && t + 1 < T_) {
        int e0 = (blk * NTHR + tid) * 8;         /* 8192 thr * 8 = B*H */
        int m = e0 >> 9, k = e0 & (H_ - 1);
        const float* src = H0in + (size_t)m * (T_ * H_) + (size_t)(t + 1) * H_ + k;
        unsigned short* dst = h0I + (size_t)(1 - p) * (B_ * H_) + e0;
#pragma unroll
        for (int j = 0; j < 8; ++j) dst[j] = f2bf(src[j]);
      }
      gbar(barcnt, bargen);
    }
  }
}

extern "C" void kernel_launch(void* const* d_in, const int* in_sizes, int n_in,
                              void* d_out, int out_size, void* d_ws, size_t ws_size,
                              hipStream_t stream)
{
  const float* H0in = (const float*)d_in[0];
  const float* Wih  = (const float*)d_in[1];
  const float* Whh  = (const float*)d_in[2];
  const float* bih  = (const float*)d_in[3];
  const float* bhh  = (const float*)d_in[4];
  float* out = (float*)d_out;
  (void)in_sizes; (void)n_in; (void)out_size;

  char* ws = (char*)d_ws;
  size_t off = 0;
  auto alloc = [&](size_t bytes) -> char* {
    char* ptr = ws + off;
    off += bytes;
    off = (off + 255) & ~(size_t)255;
    return ptr;
  };
  unsigned* barcnt = (unsigned*)alloc(8);
  unsigned* bargen = barcnt + 1;
  float* bias = (float*)alloc((size_t)4 * G_ * sizeof(float));
  float* m0s  = (float*)alloc((size_t)B_ * H_ * sizeof(float));
  float* m1s  = (float*)alloc((size_t)L_ * B_ * H_ * sizeof(float));
  unsigned short* h0I    = (unsigned short*)alloc((size_t)2 * B_ * H_ * 2);
  unsigned short* h0_mid = (unsigned short*)alloc((size_t)B_ * H_ * 2);
  unsigned short* h0_fin = (unsigned short*)alloc((size_t)B_ * H_ * 2);
  unsigned short* h1b    = (unsigned short*)alloc((size_t)4 * B_ * H_ * 2);
  unsigned short* Wf     = (unsigned short*)alloc(WF_TOTAL * 2);
  const bool pre = (ws_size >= off);   /* packed weights fit in scratch? */

  grid_lstm_prep<<<dim3(1024), dim3(256), 0, stream>>>(
      H0in, Wih, Whh, bih, bhh, bias, Wf, m1s, h1b, h0I, barcnt, pre ? 1 : 0);
  if (pre) {
    grid_lstm_main<true><<<dim3(NBLK), dim3(NTHR), 0, stream>>>(
        H0in, Wih, Whh, out, bias, Wf, m0s, m1s, h0I, h0_mid, h0_fin, h1b, barcnt, bargen);
  } else {
    grid_lstm_main<false><<<dim3(NBLK), dim3(NTHR), 0, stream>>>(
        H0in, Wih, Whh, out, bias, Wf, m0s, m1s, h0I, h0_mid, h0_fin, h1b, barcnt, bargen);
  }
}
